// BertSelfAttention_26877905339251
// MI455X (gfx1250) — compile-verified, matrix operand order checked by reading, not run on hardware
//
#include <hip/hip_runtime.h>

// ---------------------------------------------------------------------------
// BERT self-attention for MI455X (gfx1250, wave32, WMMA bf16, async-to-LDS).
//   B=8, S=1024, H=1024, NH=16, HD=64
// Kernel A: pack-convert X fp32 -> bf16
// Kernel B: convert + transpose Wq/Wk/Wv -> bf16 Wt[n][k]
// Kernel 1: QKV projection GEMM, tile 256(M)x64(N), 8 waves x 32-row strips,
//           double-buffered LDS filled via global_load_async_to_lds_b128.
//           Q,K stored bf16 [bh][s][d]; V stored transposed [bh][d][s].
// Kernel 2: flash attention, K/V tiles async-staged to double-buffered LDS
//           shared by all 4 waves; online softmax; O = P*V.
// ---------------------------------------------------------------------------

typedef __attribute__((ext_vector_type(16))) __bf16 v16bf;
typedef __attribute__((ext_vector_type(8)))  __bf16 v8bf;
typedef __attribute__((ext_vector_type(8)))  float  v8f;

#define BATCH 8
#define SEQ   1024
#define HID   1024
#define NH    16
#define HD    64

__device__ __forceinline__ __bf16 to_bf16(float f) {
  unsigned u = __builtin_bit_cast(unsigned, f);
  u += 0x7FFFu + ((u >> 16) & 1u);                 // round-to-nearest-even
  unsigned short h = (unsigned short)(u >> 16);
  return __builtin_bit_cast(__bf16, h);
}

__device__ __forceinline__ unsigned pack_bf16x2(float a, float b) {
  unsigned ua = __builtin_bit_cast(unsigned, a);
  unsigned ub = __builtin_bit_cast(unsigned, b);
  ua += 0x7FFFu + ((ua >> 16) & 1u);
  ub += 0x7FFFu + ((ub >> 16) & 1u);
  return (ua >> 16) | (ub & 0xFFFF0000u);
}

// 16B-aligned vector loads so the backend emits b128 loads.
__device__ __forceinline__ v16bf load_frag(const __bf16* p0, const __bf16* p1) {
  v8bf lo = *(const v8bf*)p0;
  v8bf hi = *(const v8bf*)p1;
  return __builtin_shufflevector(lo, hi, 0,1,2,3,4,5,6,7,8,9,10,11,12,13,14,15);
}

__device__ __forceinline__ v8f v8f_zero() {
  v8f z;
  #pragma unroll
  for (int i = 0; i < 8; ++i) z[i] = 0.0f;
  return z;
}

// CDNA5 async copy: 16 bytes global -> LDS, tracked by ASYNCcnt.
__device__ __forceinline__ void async_copy_b128(__bf16* lds, const __bf16* g) {
  unsigned loff = (unsigned)(size_t)lds;                 // low 32b = LDS offset
  unsigned long long ga = (unsigned long long)(size_t)g;
  asm volatile("global_load_async_to_lds_b128 %0, %1, off"
               :: "v"(loff), "v"(ga) : "memory");
}
__device__ __forceinline__ void wait_async0() {
  asm volatile("s_wait_asynccnt 0" ::: "memory");
}

// ---------------------------------------------------------------------------
// Kernel A: X fp32 -> bf16 row-major, 8 elements/thread, b128 stores.
// ---------------------------------------------------------------------------
__global__ __launch_bounds__(256) void convert_x_kernel(
    const float* __restrict__ in, __bf16* __restrict__ out)
{
  const size_t i0 = ((size_t)blockIdx.x * 256 + threadIdx.x) * 8;
  float4 a = *(const float4*)(in + i0);
  float4 b = *(const float4*)(in + i0 + 4);
  uint4 p;
  p.x = pack_bf16x2(a.x, a.y);
  p.y = pack_bf16x2(a.z, a.w);
  p.z = pack_bf16x2(b.x, b.y);
  p.w = pack_bf16x2(b.z, b.w);
  *(uint4*)(out + i0) = p;
}

// ---------------------------------------------------------------------------
// Kernel B: W[k][n] fp32 -> Wt[z][n][k] bf16 (transposed via LDS tile).
// ---------------------------------------------------------------------------
__global__ __launch_bounds__(256) void convert_w_kernel(
    const float* __restrict__ Wq, const float* __restrict__ Wk,
    const float* __restrict__ Wv, __bf16* __restrict__ Wt)
{
  __shared__ float tile[32][33];
  const int z  = blockIdx.z;
  const float* W = (z == 0) ? Wq : (z == 1) ? Wk : Wv;
  const int k0 = blockIdx.y * 32, n0 = blockIdx.x * 32;
  const int tx = threadIdx.x & 31, ty = threadIdx.x >> 5;   // 32 x 8

  #pragma unroll
  for (int i = 0; i < 32; i += 8)
    tile[ty + i][tx] = W[(size_t)(k0 + ty + i) * HID + (n0 + tx)];
  __syncthreads();

  __bf16* o = Wt + (size_t)z * HID * HID;
  #pragma unroll
  for (int i = 0; i < 32; i += 8)
    o[(size_t)(n0 + ty + i) * HID + (k0 + tx)] = to_bf16(tile[tx][ty + i]);
}

// ---------------------------------------------------------------------------
// Kernel 1: QKV projection. Block = 256 threads (8 waves), tile 256(M)x64(N),
// each wave owns a 32x64 strip (B-fragments reused across two A-fragments).
// ---------------------------------------------------------------------------
__device__ __forceinline__ void stage_panels(
    int tid, int buf, int k0, int mbase, int nbase,
    const __bf16* __restrict__ Xb, const __bf16* __restrict__ Wz,
    __bf16 (&Xs)[2][256][40], __bf16 (&WsT)[2][64][40])
{
  // X panel: 256 rows x 64B = 1024 x B128 chunks (4 per thread)
  #pragma unroll
  for (int i = 0; i < 4; ++i) {
    const int id = i * 256 + tid;
    const int r = id >> 2, c = (id & 3) * 8;
    async_copy_b128(&Xs[buf][r][c], Xb + (size_t)(mbase + r) * HID + k0 + c);
  }
  // W panel (pre-transposed in global): 64 rows x 64B = 256 x B128 chunks
  {
    const int r = tid >> 2, c = (tid & 3) * 8;
    async_copy_b128(&WsT[buf][r][c], Wz + (size_t)(nbase + r) * HID + k0 + c);
  }
}

__global__ __launch_bounds__(256) void qkv_proj_kernel(
    const __bf16* __restrict__ Xb, const __bf16* __restrict__ Wt,
    const float* __restrict__ bq, const float* __restrict__ bk,
    const float* __restrict__ bv,
    __bf16* __restrict__ Qw, __bf16* __restrict__ Kw, __bf16* __restrict__ Vt)
{
  __shared__ __align__(16) __bf16 Xs[2][256][40];   // 40 KB: M x Kpanel (+pad)
  __shared__ __align__(16) __bf16 WsT[2][64][40];   // 10 KB: N x Kpanel (+pad)

  const int tid   = threadIdx.x;
  const int lane  = tid & 31;
  const int wv    = tid >> 5;            // 0..7 : 32-row strip
  const int nbase = blockIdx.x * 64;
  const int mbase = blockIdx.y * 256;
  const int z     = blockIdx.z;          // 0=Q 1=K 2=V

  const __bf16* Wz   = Wt + (size_t)z * HID * HID;
  const float*  bias = (z == 0) ? bq : (z == 1) ? bk : bv;

  const int mm   = lane & 15;            // row within 16x16 tile (A) / col (B)
  const int kb   = (lane >> 4) * 8;      // K sub-offset per lane half
  const int half = lane >> 4;

  v8f acc0[4], acc1[4];
  #pragma unroll
  for (int i = 0; i < 4; ++i) { acc0[i] = v8f_zero(); acc1[i] = v8f_zero(); }

  // Prime buffer 0, then double-buffer: issue next panel, compute current.
  stage_panels(tid, 0, 0, mbase, nbase, Xb, Wz, Xs, WsT);
  wait_async0();
  __syncthreads();

  for (int k0 = 0; k0 < HID; k0 += 32) {
    const int cur = (k0 >> 5) & 1;
    if (k0 + 32 < HID)
      stage_panels(tid, cur ^ 1, k0 + 32, mbase, nbase, Xb, Wz, Xs, WsT);

    const __bf16* xr0 = &Xs[cur][wv * 32 + mm][0];
    const __bf16* xr1 = &Xs[cur][wv * 32 + 16 + mm][0];
    v16bf a0 = load_frag(xr0 + kb, xr0 + 16 + kb);
    v16bf a1 = load_frag(xr1 + kb, xr1 + 16 + kb);

    v16bf bm[4];
    #pragma unroll
    for (int nt = 0; nt < 4; ++nt) {
      const __bf16* wrow = &WsT[cur][nt * 16 + mm][0];
      bm[nt] = load_frag(wrow + kb, wrow + 16 + kb);
    }
    #pragma unroll
    for (int nt = 0; nt < 4; ++nt) {
      acc0[nt] = __builtin_amdgcn_wmma_f32_16x16x32_bf16(
          false, a0, false, bm[nt], (short)0, acc0[nt], false, false);
      acc1[nt] = __builtin_amdgcn_wmma_f32_16x16x32_bf16(
          false, a1, false, bm[nt], (short)0, acc1[nt], false, false);
    }

    wait_async0();       // this wave's async fills of the next buffer done
    __syncthreads();     // everyone's fills done before next iteration reads
  }

  // Epilogue: bias add, strided bf16 store (Q/K: [s][d], V: transposed [d][s]).
  __bf16* outp = (z == 0) ? Qw : (z == 1) ? Kw : Vt;
  const int sStr = (z == 2) ? 1 : HD;
  const int dStr = (z == 2) ? SEQ : 1;

  #pragma unroll
  for (int p = 0; p < 2; ++p) {
    #pragma unroll
    for (int nt = 0; nt < 4; ++nt) {
      const int n  = nbase + nt * 16 + mm;
      const float bval = bias[n];
      const int h = n >> 6, d = n & 63;
      #pragma unroll
      for (int r = 0; r < 8; ++r) {
        const int m = mbase + wv * 32 + p * 16 + r + 8 * half;  // global row
        const int bb = m >> 10, s = m & 1023;
        const float val = (p ? acc1[nt][r] : acc0[nt][r]) + bval;
        const size_t base = (size_t)(bb * NH + h) * (SEQ * HD);
        outp[base + (size_t)s * sStr + (size_t)d * dStr] = to_bf16(val);
      }
    }
  }
}

// ---------------------------------------------------------------------------
// Kernel 2: flash attention. Block = 128 threads (4 waves), each wave owns a
// 16x64 Q strip. K/V tiles are async-staged into double-buffered LDS shared
// by the whole block; P goes C-layout -> LDS -> A-layout for the PV matmul.
// ---------------------------------------------------------------------------
__device__ __forceinline__ void stage_kv(
    int tid, int buf, int kt,
    const __bf16* __restrict__ K, const __bf16* __restrict__ V,
    __bf16 (&Ks)[2][64][72], __bf16 (&Vs)[2][64][72])
{
  // K tile: 64 keys x 128B  |  V tile: 64 d-rows x 128B  (4+4 chunks/thread)
  #pragma unroll
  for (int i = 0; i < 4; ++i) {
    const int id = i * 128 + tid;
    const int r = id >> 3, c = (id & 7) * 8;
    async_copy_b128(&Ks[buf][r][c], K + (size_t)(kt * 64 + r) * HD + c);
    async_copy_b128(&Vs[buf][r][c], V + (size_t)r * SEQ + kt * 64 + c);
  }
}

__global__ __launch_bounds__(128) void attention_kernel(
    const __bf16* __restrict__ Qw, const __bf16* __restrict__ Kw,
    const __bf16* __restrict__ Vt, const float* __restrict__ mask,
    float* __restrict__ out)
{
  __shared__ __align__(16) __bf16 Ks[2][64][72];     // 18 KB
  __shared__ __align__(16) __bf16 Vs[2][64][72];     // 18 KB
  __shared__ __align__(16) __bf16 Plds[4][16][72];   //  9 KB per-wave P tile

  const int tid  = threadIdx.x;
  const int lane = tid & 31;
  const int wv   = tid >> 5;                   // 0..3
  const int bh   = blockIdx.x;                 // 0..127
  const int b    = bh >> 4, h = bh & 15;
  const int q0   = blockIdx.y * 64 + wv * 16;  // this wave's first q row

  const __bf16* Q = Qw + (size_t)bh * SEQ * HD;
  const __bf16* K = Kw + (size_t)bh * SEQ * HD;
  const __bf16* V = Vt + (size_t)bh * SEQ * HD;   // [d][s]

  const int mm   = lane & 15;
  const int kb   = (lane >> 4) * 8;
  const int half = lane >> 4;

  // Q strip as two A-fragments (d = 0..31, 32..63), resident all kernel.
  const __bf16* qrow = Q + (size_t)(q0 + mm) * HD;
  v16bf aq0 = load_frag(qrow + kb,      qrow + 16 + kb);
  v16bf aq1 = load_frag(qrow + 32 + kb, qrow + 48 + kb);

  v8f O[4];
  float m_run[8], l_run[8];
  #pragma unroll
  for (int i = 0; i < 4; ++i) O[i] = v8f_zero();
  #pragma unroll
  for (int r = 0; r < 8; ++r) { m_run[r] = -3.0e38f; l_run[r] = 0.0f; }

  stage_kv(tid, 0, 0, K, V, Ks, Vs);
  wait_async0();
  __syncthreads();

  for (int kt = 0; kt < 16; ++kt) {
    const int cur = kt & 1;
    if (kt + 1 < 16) stage_kv(tid, cur ^ 1, kt + 1, K, V, Ks, Vs);

    // ---- scores: S = Q K^T  (contraction over d = 64, two WMMA k-steps)
    v8f sacc[4];
    #pragma unroll
    for (int nt = 0; nt < 4; ++nt) {
      sacc[nt] = v8f_zero();
      const __bf16* kr = &Ks[cur][nt * 16 + mm][0];
      v16bf b0 = load_frag(kr + kb,      kr + 16 + kb);
      v16bf b1 = load_frag(kr + 32 + kb, kr + 48 + kb);
      sacc[nt] = __builtin_amdgcn_wmma_f32_16x16x32_bf16(
          false, aq0, false, b0, (short)0, sacc[nt], false, false);
      sacc[nt] = __builtin_amdgcn_wmma_f32_16x16x32_bf16(
          false, aq1, false, b1, (short)0, sacc[nt], false, false);
    }

    // ---- scale + additive mask
    #pragma unroll
    for (int nt = 0; nt < 4; ++nt) {
      const float mk = mask[b * SEQ + kt * 64 + nt * 16 + mm];
      #pragma unroll
      for (int r = 0; r < 8; ++r) sacc[nt][r] = sacc[nt][r] * 0.125f + mk;
    }

    // ---- online softmax: row stats live in the 16-lane half owning the row
    float tmax[8];
    #pragma unroll
    for (int r = 0; r < 8; ++r) {
      float t = fmaxf(fmaxf(sacc[0][r], sacc[1][r]),
                      fmaxf(sacc[2][r], sacc[3][r]));
      #pragma unroll
      for (int off = 1; off < 16; off <<= 1)
        t = fmaxf(t, __shfl_xor(t, off, 32));
      tmax[r] = t;
    }
    float alpha[8], rsum[8];
    #pragma unroll
    for (int r = 0; r < 8; ++r) {
      const float mnew = fmaxf(m_run[r], tmax[r]);
      alpha[r] = __expf(m_run[r] - mnew);
      m_run[r] = mnew;
      rsum[r]  = 0.0f;
    }
    #pragma unroll
    for (int nt = 0; nt < 4; ++nt)
      #pragma unroll
      for (int r = 0; r < 8; ++r) {
        const float p = __expf(sacc[nt][r] - m_run[r]);
        sacc[nt][r] = p;
        rsum[r] += p;
      }
    #pragma unroll
    for (int r = 0; r < 8; ++r) {
      float t = rsum[r];
      #pragma unroll
      for (int off = 1; off < 16; off <<= 1) t += __shfl_xor(t, off, 32);
      l_run[r] = l_run[r] * alpha[r] + t;
    }
    #pragma unroll
    for (int nt = 0; nt < 4; ++nt)
      #pragma unroll
      for (int r = 0; r < 8; ++r) O[nt][r] *= alpha[r];

    // ---- P: C-layout -> LDS (bf16) -> A-layout fragments
    #pragma unroll
    for (int nt = 0; nt < 4; ++nt)
      #pragma unroll
      for (int r = 0; r < 8; ++r)
        Plds[wv][r + 8 * half][nt * 16 + mm] = to_bf16(sacc[nt][r]);
    __builtin_amdgcn_wave_barrier();   // intra-wave: DS ops are in-order

    const __bf16* prow = &Plds[wv][mm][0];
    v16bf ap0 = load_frag(prow + kb,      prow + 16 + kb);
    v16bf ap1 = load_frag(prow + 32 + kb, prow + 48 + kb);

    // ---- O += P V  (V rows are head-dims; columns contiguous over keys)
    #pragma unroll
    for (int nt = 0; nt < 4; ++nt) {
      const __bf16* vc = &Vs[cur][nt * 16 + mm][0];
      v16bf bv0 = load_frag(vc + kb,      vc + 16 + kb);
      v16bf bv1 = load_frag(vc + 32 + kb, vc + 48 + kb);
      O[nt] = __builtin_amdgcn_wmma_f32_16x16x32_bf16(
          false, ap0, false, bv0, (short)0, O[nt], false, false);
      O[nt] = __builtin_amdgcn_wmma_f32_16x16x32_bf16(
          false, ap1, false, bv1, (short)0, O[nt], false, false);
    }

    wait_async0();       // this wave's fills of the next K/V buffer done
    __syncthreads();     // all waves done reading cur before it is refilled
  }

  // ---- finalize: divide by softmax sum, merged-heads fp32 output
  #pragma unroll
  for (int nt = 0; nt < 4; ++nt) {
    const int n = h * 64 + nt * 16 + mm;
    #pragma unroll
    for (int r = 0; r < 8; ++r) {
      const int s = q0 + r + 8 * half;
      out[(size_t)(b * SEQ + s) * HID + n] = O[nt][r] / l_run[r];
    }
  }
}

// ---------------------------------------------------------------------------
extern "C" void kernel_launch(void* const* d_in, const int* in_sizes, int n_in,
                              void* d_out, int out_size, void* d_ws, size_t ws_size,
                              hipStream_t stream) {
  const float* X    = (const float*)d_in[0];
  const float* mask = (const float*)d_in[1];
  const float* Wq   = (const float*)d_in[2];
  const float* bq   = (const float*)d_in[3];
  const float* Wk   = (const float*)d_in[4];
  const float* bk   = (const float*)d_in[5];
  const float* Wv   = (const float*)d_in[6];
  const float* bv   = (const float*)d_in[7];

  // Workspace (bf16): Q, K, V^T (3 x 16 MB) + Xb (16 MB) + Wt (6 MB) ~ 70 MB
  const size_t per = (size_t)BATCH * NH * SEQ * HD;   // 8,388,608 elements
  __bf16* Qw = (__bf16*)d_ws;
  __bf16* Kw = Qw + per;
  __bf16* Vt = Kw + per;
  __bf16* Xb = Vt + per;                 // [B*S][H] bf16
  __bf16* Wt = Xb + per;                 // [3][n=H][k=H] bf16 (transposed)

  convert_x_kernel<<<(BATCH * SEQ * HID) / (256 * 8), 256, 0, stream>>>(X, Xb);
  convert_w_kernel<<<dim3(HID / 32, HID / 32, 3), 256, 0, stream>>>(Wq, Wk, Wv, Wt);

  qkv_proj_kernel<<<dim3(HID / 64, (BATCH * SEQ) / 256, 3), 256, 0, stream>>>(
      Xb, Wt, bq, bk, bv, Qw, Kw, Vt);

  attention_kernel<<<dim3(BATCH * NH, SEQ / 64), 128, 0, stream>>>(
      Qw, Kw, Vt, mask, (float*)d_out);
}
